// OrcDeBERTa_3959959847306
// MI455X (gfx1250) — matrix operand order repaired and checked by reading, not compiled, not run on hardware
//
#include <hip/hip_runtime.h>
#include <hip/hip_bf16.h>
#include <math.h>

// ---------------------------------------------------------------------------
// CDNA5 (gfx1250) DeBERTa forward. wave32. All GEMMs via v_wmma_f32_16x16x32_bf16.
// Weights pre-transposed to [N][K] bf16 so both staging sides are b128 vector
// ops; double-buffered K pipeline; all 8 B fragments preloaded so WMMAs issue
// back-to-back with staggered dscnt waits.
// ---------------------------------------------------------------------------

typedef __attribute__((ext_vector_type(16))) __bf16 v16bf;
typedef __attribute__((ext_vector_type(8)))  __bf16 v8bf;
typedef __attribute__((ext_vector_type(8)))  float  v8f;

union F16 { v16bf v; v8bf h[2]; };

#define BATCH 64
#define SEQ   300
#define NHEAD 4
#define DHEAD 32
#define DMODEL 128
#define NLAYER 8
#define MTOK  (BATCH * SEQ)       // 19200 rows
#define INV_SCALE 0.1020620726159658f   // 1/sqrt(96)

__device__ __forceinline__ __bf16 f2bf(float f) {
  unsigned u = __builtin_bit_cast(unsigned, f);
  unsigned r = u + 0x7FFFu + ((u >> 16) & 1u);       // round-to-nearest-even
  return __builtin_bit_cast(__bf16, (unsigned short)(r >> 16));
}
__device__ __forceinline__ float gelu_f(float x) {
  return 0.5f * x * (1.0f + erff(x * 0.70710678118654752f));
}

// ======================= generic bf16 WMMA GEMM ============================
// Block = 256 threads (8 waves). Tile: 128 rows x 128 cols. K chunks of 32.
// Wt is the TRANSPOSED weight: Wt[n][k], row stride ldw (== K).
// flags: 1 = add bias, 2 = GELU, 4 = residual + LayerNorm (N==128, gridDim.y==1)
#define GKC 32

__global__ __launch_bounds__(256)
void gemm_bf16_kernel(const __bf16* __restrict__ A, int lda,
                      const __bf16* __restrict__ Wt, int ldw,
                      const float* __restrict__ bias,
                      const float* __restrict__ resid, int ldr,
                      const float* __restrict__ ln_s, const float* __restrict__ ln_b,
                      float* __restrict__ outF, int ldoF,
                      __bf16* __restrict__ outB, int ldoB,
                      int M, int N, int K, int flags) {
  // Both stages fragment-contiguous: ldsBt[buf][n][k], ldsA[buf][w][m][k].
  __shared__ alignas(64) __bf16 ldsBt[2][128][GKC];     // 16 KB
  __shared__ alignas(64) __bf16 ldsA[2][8][16][GKC];    // 16 KB

  const int tid   = threadIdx.x;
  const int lane  = tid & 31;
  const int wave  = tid >> 5;
  const int m0    = blockIdx.x * 128;
  const int n0    = blockIdx.y * 128;
  const int col16 = lane & 15;
  const int khalf = (lane >> 4) & 1;

  // staging decode (fixed per thread): both global and LDS sides contiguous
  const int bnn = tid >> 1;              // B: output column 0..127
  const int bkh = (tid & 1) * 16;        // B: 16 consecutive k
  const int ar  = lane >> 1;             // A: row 0..15
  const int ah  = (lane & 1) * 16;       // A: 16 consecutive k

  __bf16 bReg[16], aReg[16];

  auto fetch = [&](int kc) {
    const int kbase = kc * GKC;
    {
      int gn = n0 + bnn;
      if (gn < N && kbase + bkh + 15 < K) {
        const __bf16* src = &Wt[(size_t)gn * ldw + kbase + bkh];
#pragma unroll
        for (int i = 0; i < 16; ++i) bReg[i] = src[i];
        if (kbase + bkh + 2 * GKC < K)
          __builtin_prefetch(src + 2 * GKC, 0, 0);      // global_prefetch_b8
      } else {
#pragma unroll
        for (int i = 0; i < 16; ++i) {
          int gk = kbase + bkh + i;
          bReg[i] = (gn < N && gk < K) ? Wt[(size_t)gn * ldw + gk] : f2bf(0.f);
        }
      }
    }
    {
      int gr = m0 + wave * 16 + ar;
      if (gr < M && kbase + ah + 15 < K) {
        const __bf16* src = &A[(size_t)gr * lda + kbase + ah];
#pragma unroll
        for (int i = 0; i < 16; ++i) aReg[i] = src[i];
      } else {
#pragma unroll
        for (int i = 0; i < 16; ++i) {
          int gk = kbase + ah + i;
          aReg[i] = (gr < M && gk < K) ? A[(size_t)gr * lda + gk] : f2bf(0.f);
        }
      }
    }
  };
  auto stash = [&](int s) {
#pragma unroll
    for (int i = 0; i < 16; ++i) ldsBt[s][bnn][bkh + i] = bReg[i];      // b128 x2
#pragma unroll
    for (int i = 0; i < 16; ++i) ldsA[s][wave][ar][ah + i] = aReg[i];   // b128 x2
  };

  v8f acc[8] = {};
  const int nK = (K + GKC - 1) / GKC;

  fetch(0);
  stash(0);
  int buf = 0;
  for (int kc = 0; kc < nK; ++kc) {
    __syncthreads();                 // stage `buf` ready for all waves
    if (kc + 1 < nK) fetch(kc + 1); // overlap next-chunk global loads with WMMAs

    // Preload A fragment and ALL 8 B fragments (all live -> loads cannot be
    // sunk; WMMAs then issue back-to-back with staggered dscnt waits).
    F16 af, bfr[8];
    af.v = *(const v16bf*)&ldsA[buf][wave][col16][khalf * 16];
#pragma unroll
    for (int t = 0; t < 8; ++t)
      bfr[t].v = *(const v16bf*)&ldsBt[buf][t * 16 + col16][khalf * 16];
#pragma unroll
    for (int t = 0; t < 8; ++t)
      acc[t] = __builtin_amdgcn_wmma_f32_16x16x32_bf16(
          false, af.v, false, bfr[t].v, (short)0, acc[t], false, false);

    if (kc + 1 < nK) stash(buf ^ 1); // writers touch other stage: no extra barrier
    buf ^= 1;
  }

  // ======================= epilogue =======================
  if (flags & 4) {
    float val[8][8];
#pragma unroll
    for (int t = 0; t < 8; ++t)
#pragma unroll
      for (int r = 0; r < 8; ++r) {
        int m = m0 + wave * 16 + r + 8 * khalf;
        int n = n0 + t * 16 + col16;
        float v = acc[t][r];
        if (flags & 1) v += bias[n];
        if (m < M) v += resid[m * ldr + n];
        val[t][r] = v;
      }
#pragma unroll
    for (int r = 0; r < 8; ++r) {
      float s = 0.f, q = 0.f;
#pragma unroll
      for (int t = 0; t < 8; ++t) { s += val[t][r]; q += val[t][r] * val[t][r]; }
#pragma unroll
      for (int o = 1; o < 16; o <<= 1) {
        s += __shfl_xor(s, o, 32);
        q += __shfl_xor(q, o, 32);
      }
      float mean = s * (1.f / 128.f);
      float var  = q * (1.f / 128.f) - mean * mean;
      float inv  = rsqrtf(var + 1e-7f);
      int m = m0 + wave * 16 + r + 8 * khalf;
      if (m < M) {
#pragma unroll
        for (int t = 0; t < 8; ++t) {
          int n = n0 + t * 16 + col16;
          float o2 = (val[t][r] - mean) * inv * ln_s[n] + ln_b[n];
          if (outF) outF[m * ldoF + n] = o2;
          if (outB) outB[m * ldoB + n] = f2bf(o2);
        }
      }
    }
  } else {
#pragma unroll
    for (int t = 0; t < 8; ++t)
#pragma unroll
      for (int r = 0; r < 8; ++r) {
        int m = m0 + wave * 16 + r + 8 * khalf;
        int n = n0 + t * 16 + col16;
        if (m < M && n < N) {
          float v = acc[t][r];
          if (flags & 1) v += bias[n];
          if (flags & 2) v = gelu_f(v);
          if (outF) outF[m * ldoF + n] = v;
          if (outB) outB[m * ldoB + n] = f2bf(v);
        }
      }
  }
}

// ======================= fused disentangled attention ======================
// One wave per (16-query tile, b, h). qkv: [B*S, 384] f32.
__global__ __launch_bounds__(32)
void attn_kernel(const float* __restrict__ qkv,
                 const float* __restrict__ qb, const float* __restrict__ vb,
                 const __bf16* __restrict__ posk, const __bf16* __restrict__ posq,
                 const float* __restrict__ mask, __bf16* __restrict__ ctx) {
  __shared__ alignas(64) float  sc[16][320];     // 20480 B (reused as vt after softmax)
  __shared__ alignas(64) __bf16 pr[16][320];     // 10240 B
  __shared__ alignas(64) __bf16 ldsQ[16][32];    // 1 KB
  __shared__ alignas(64) __bf16 ldsK[16][32];    // 1 KB
  __shared__ float c2p[16][16];
  __shared__ float p2c[16][16];

  const int lane  = threadIdx.x & 31;
  const int col16 = lane & 15;
  const int khalf = (lane >> 4) & 1;
  const int b     = blockIdx.y >> 2;
  const int h     = blockIdx.y & 3;
  const int q0    = blockIdx.x * 16;
  const float* maskb = mask + b * SEQ;

  // ---- stage q tile (bias + 1/sqrt(96)) into fragment-contiguous LDS ----
  {
    int rr = lane >> 1;
    int hb = (lane & 1) * 16;
    int qr = q0 + rr;
    const float* src = &qkv[(size_t)(b * SEQ + (qr < SEQ ? qr : 0)) * 384 + h * 96 + hb];
#pragma unroll
    for (int i = 0; i < 16; ++i) {
      float x = (qr < SEQ) ? (src[i] + qb[h * 32 + hb + i]) * INV_SCALE : 0.f;
      ldsQ[rr][hb + i] = f2bf(x);
    }
  }
  __syncthreads();

  F16 qf;
  qf.v = *(const v16bf*)&ldsQ[col16][khalf * 16];

  // ---- pos_k / pos_q fragments straight from global (layout is contiguous) ----
  F16 pkf, pqf;
  pkf.v = *(const v16bf*)&posk[h * 512 + col16 * 32 + khalf * 16];
  pqf.v = *(const v16bf*)&posq[h * 512 + col16 * 32 + khalf * 16];

  // ---- c2p tile = q @ pos_k^T ----
  {
    v8f z = {};
    v8f cp = __builtin_amdgcn_wmma_f32_16x16x32_bf16(false, qf.v, false, pkf.v,
                                                     (short)0, z, false, false);
#pragma unroll
    for (int r = 0; r < 8; ++r) c2p[r + 8 * khalf][col16] = cp[r];
  }
  __syncthreads();

  // ---- scores + p2c per 16-key tile ----
  for (int kb = 0; kb < 19; ++kb) {
    {
      int rr = lane >> 1;
      int hb = (lane & 1) * 16;
      int kr = kb * 16 + rr;
      const float* src =
          &qkv[(size_t)(b * SEQ + (kr < SEQ ? kr : 0)) * 384 + h * 96 + 32 + hb];
#pragma unroll
      for (int i = 0; i < 16; ++i)
        ldsK[rr][hb + i] = f2bf((kr < SEQ) ? src[i] : 0.f);
    }
    __syncthreads();

    F16 kB, kA;                 // same staging serves both operand layouts
    kB.v    = *(const v16bf*)&ldsK[col16][khalf * 16];
    kA.v    = *(const v16bf*)&ldsK[col16][khalf * 16];

    v8f z1 = {}, z2 = {};
    v8f sa = __builtin_amdgcn_wmma_f32_16x16x32_bf16(false, qf.v, false, kB.v,
                                                     (short)0, z1, false, false);
    v8f pa = __builtin_amdgcn_wmma_f32_16x16x32_bf16(false, kA.v, false, pqf.v,
                                                     (short)0, z2, false, false);
#pragma unroll
    for (int r = 0; r < 8; ++r) {
      sc[r + 8 * khalf][kb * 16 + col16] = sa[r];
      p2c[r + 8 * khalf][col16]          = pa[r];
    }
    __syncthreads();
    // disentangled bias: both terms indexed by clip(q-k+8, 0, 15)
#pragma unroll
    for (int i = 0; i < 8; ++i) {
      int e  = lane * 8 + i;
      int qi = e >> 4;
      int kj = e & 15;
      int rel = (q0 + qi) - (kb * 16 + kj);
      int idx = rel + 8; idx = idx < 0 ? 0 : (idx > 15 ? 15 : idx);
      sc[qi][kb * 16 + kj] += c2p[qi][idx] + p2c[kj][idx];
    }
    __syncthreads();
  }

  // ---- masking ----
  for (int j = lane; j < 320; j += 32) {
    float mk = (j < SEQ) ? maskb[j] : 0.f;
#pragma unroll
    for (int i = 0; i < 16; ++i) {
      int qr = q0 + i;
      float mq = (qr < SEQ) ? maskb[qr] : 0.f;
      if (!(mk > 0.f && mq > 0.f)) sc[i][j] = -1e30f;
    }
  }
  __syncthreads();
  // ---- softmax (row per lane 0-15) ----
  if (lane < 16) {
    float mx = -1e30f;
    for (int j = 0; j < 320; ++j) mx = fmaxf(mx, sc[lane][j]);
    bool valid = mx > -1e29f;
    float sum = 0.f;
    for (int j = 0; j < 320; ++j) {
      float p = valid ? expf(sc[lane][j] - mx) : 0.f;
      sc[lane][j] = p; sum += p;
    }
    float inv = (valid && sum > 0.f) ? 1.f / sum : 0.f;
    for (int j = 0; j < 320; ++j) pr[lane][j] = f2bf(sc[lane][j] * inv);
  }
  __syncthreads();

  // ---- stage v TRANSPOSED into the (now free) score scratch: vt[32][320] ----
  __bf16 (*ldsVt)[320] = reinterpret_cast<__bf16 (*)[320]>(&sc[0][0]);
  for (int key = lane; key < 320; key += 32) {
    bool ok = key < SEQ;
    const float* src = &qkv[(size_t)(b * SEQ + (ok ? key : 0)) * 384 + h * 96 + 64];
#pragma unroll
    for (int d = 0; d < 32; ++d)
      ldsVt[d][key] = f2bf(ok ? src[d] + vb[h * 32 + d] : 0.f);
  }
  __syncthreads();

  // ---- ctx = probs @ v  (K = 320, 10 chunks x 2 N-tiles) ----
  v8f cacc[2] = {};
  for (int kc = 0; kc < 10; ++kc) {
    F16 pf;
    pf.v = *(const v16bf*)&pr[col16][kc * 32 + khalf * 16];
#pragma unroll
    for (int nt = 0; nt < 2; ++nt) {
      F16 vf;
      vf.v = *(const v16bf*)&ldsVt[nt * 16 + col16][kc * 32 + khalf * 16];
      cacc[nt] = __builtin_amdgcn_wmma_f32_16x16x32_bf16(
          false, pf.v, false, vf.v, (short)0, cacc[nt], false, false);
    }
  }
#pragma unroll
  for (int nt = 0; nt < 2; ++nt)
#pragma unroll
    for (int r = 0; r < 8; ++r) {
      int qr = q0 + r + 8 * khalf;
      if (qr < SEQ)
        ctx[(size_t)(b * SEQ + qr) * 128 + h * 32 + nt * 16 + col16] = f2bf(cacc[nt][r]);
    }
}

// ======================= small helper kernels ==============================
// convert + transpose: in f32 [L][K][N] -> out bf16 [L][N][K]
__global__ void convT_kernel(const float* __restrict__ in,
                             __bf16* __restrict__ out, int K, int N, int L) {
  long i = (long)blockIdx.x * blockDim.x + threadIdx.x;
  long tot = (long)L * K * N;
  if (i >= tot) return;
  int  l   = (int)(i / ((long)K * N));
  long rem = i % ((long)K * N);
  int  n   = (int)(rem / K);
  int  k   = (int)(rem % K);
  out[(size_t)l * K * N + (size_t)n * K + k] =
      f2bf(in[(size_t)l * K * N + (size_t)k * N + n]);
}

__global__ void embed_first_kernel(const float* __restrict__ inp,
                                   const float* __restrict__ W1,
                                   const float* __restrict__ b1,
                                   __bf16* __restrict__ e1) {
  int t = blockIdx.x * blockDim.x + threadIdx.x;
  if (t >= MTOK * 64) return;
  int j = t & 63, tok = t >> 6;
  const float* in = inp + (size_t)tok * 4;
  float s = b1[j] + in[0] * W1[j] + in[1] * W1[64 + j] + in[2] * W1[128 + j];
  e1[t] = f2bf(gelu_f(s));
}

__global__ void embed_ln_kernel(const float* __restrict__ e3,
                                const float* __restrict__ inp,
                                const float* __restrict__ tokemb,
                                const float* __restrict__ lns,
                                const float* __restrict__ lnb,
                                const float* __restrict__ mask,
                                float* __restrict__ xf, __bf16* __restrict__ xb) {
  int t = blockIdx.x * blockDim.x + threadIdx.x;
  if (t >= MTOK) return;
  int tok = (int)inp[(size_t)t * 4 + 3];
  const float* te = tokemb + tok * DMODEL;
  const float* e  = e3 + (size_t)t * DMODEL;
  float mu = 0.f;
  for (int n = 0; n < DMODEL; ++n) mu += e[n] + te[n];
  mu *= (1.f / DMODEL);
  float var = 0.f;
  for (int n = 0; n < DMODEL; ++n) { float d = e[n] + te[n] - mu; var += d * d; }
  var *= (1.f / DMODEL);
  float inv = rsqrtf(var + 1e-7f);
  float m = mask[t];
  for (int n = 0; n < DMODEL; ++n) {
    float o = ((e[n] + te[n] - mu) * inv * lns[n] + lnb[n]) * m;
    xf[(size_t)t * DMODEL + n] = o;
    xb[(size_t)t * DMODEL + n] = f2bf(o);
  }
}

__global__ void posqk_kernel(const float* __restrict__ rel,
                             const float* __restrict__ pkW,
                             const float* __restrict__ pqW,
                             const float* __restrict__ pqb,
                             __bf16* __restrict__ posk, __bf16* __restrict__ posq) {
  int t = blockIdx.x * blockDim.x + threadIdx.x;
  if (t >= NLAYER * 2 * 16 * DMODEL) return;
  int col   = t & 127;
  int r     = (t >> 7) & 15;
  int which = (t >> 11) & 1;
  int l     = t >> 12;
  const float* W = (which ? pqW : pkW) + (size_t)l * DMODEL * DMODEL;
  float s = 0.f;
  for (int k = 0; k < DMODEL; ++k) s += rel[r * DMODEL + k] * W[k * DMODEL + col];
  int h = col >> 5, d = col & 31;
  size_t off = (size_t)l * 4096 + h * 512 + r * 32 + d;
  if (which) posq[off] = f2bf((s + pqb[l * DMODEL + col]) * INV_SCALE);
  else       posk[off] = f2bf(s);
}

// ======================= host orchestration ================================
extern "C" void kernel_launch(void* const* d_in, const int* in_sizes, int n_in,
                              void* d_out, int out_size, void* d_ws, size_t ws_size,
                              hipStream_t stream) {
  (void)in_sizes; (void)n_in; (void)out_size; (void)ws_size;
  const float* inputs     = (const float*)d_in[0];
  const float* mask       = (const float*)d_in[1];
  const float* emb_W1     = (const float*)d_in[2];
  const float* emb_b1     = (const float*)d_in[3];
  const float* emb_W2     = (const float*)d_in[4];
  const float* emb_b2     = (const float*)d_in[5];
  const float* emb_W3     = (const float*)d_in[6];
  const float* emb_b3     = (const float*)d_in[7];
  const float* tok_emb    = (const float*)d_in[8];
  const float* emb_ln_s   = (const float*)d_in[9];
  const float* emb_ln_b   = (const float*)d_in[10];
  const float* rel_emb    = (const float*)d_in[11];
  const float* qkv_W      = (const float*)d_in[12];
  const float* q_bias     = (const float*)d_in[13];
  const float* v_bias     = (const float*)d_in[14];
  const float* pos_k_W    = (const float*)d_in[15];
  const float* pos_q_W    = (const float*)d_in[16];
  const float* pos_q_b    = (const float*)d_in[17];
  const float* attn_out_W = (const float*)d_in[18];
  const float* attn_out_b = (const float*)d_in[19];
  const float* ln1_s      = (const float*)d_in[20];
  const float* ln1_b      = (const float*)d_in[21];
  const float* ffn_W1     = (const float*)d_in[22];
  const float* ffn_b1     = (const float*)d_in[23];
  const float* ffn_W2     = (const float*)d_in[24];
  const float* ffn_b2     = (const float*)d_in[25];
  const float* ln2_s      = (const float*)d_in[26];
  const float* ln2_b      = (const float*)d_in[27];
  const float* rec_W1     = (const float*)d_in[28];
  const float* rec_b1     = (const float*)d_in[29];
  const float* rec_W2     = (const float*)d_in[30];
  const float* rec_b2     = (const float*)d_in[31];
  const float* rec_W3     = (const float*)d_in[32];
  const float* rec_b3     = (const float*)d_in[33];
  const float* conn_W     = (const float*)d_in[34];
  const float* conn_b     = (const float*)d_in[35];
  float* out = (float*)d_out;

  size_t off = 0;
  auto alloc = [&](size_t bytes) -> void* {
    void* p = (char*)d_ws + off;
    off += (bytes + 255) & ~(size_t)255;
    return p;
  };
  float*  x_f32 = (float*) alloc((size_t)MTOK * 128 * 4);
  __bf16* x_bf  = (__bf16*)alloc((size_t)MTOK * 128 * 2);
  float*  qkv   = (float*) alloc((size_t)MTOK * 384 * 4);
  __bf16* ctx   = (__bf16*)alloc((size_t)MTOK * 128 * 2);
  __bf16* hmid  = (__bf16*)alloc((size_t)MTOK * 512 * 2);
  __bf16* e1    = (__bf16*)alloc((size_t)MTOK * 64 * 2);
  __bf16* e2    = (__bf16*)alloc((size_t)MTOK * 128 * 2);
  float*  e3    = (float*) alloc((size_t)MTOK * 128 * 4);
  __bf16* r1    = (__bf16*)alloc((size_t)MTOK * 128 * 2);
  __bf16* r2    = (__bf16*)alloc((size_t)MTOK * 64 * 2);
  __bf16* r3    = (__bf16*)alloc((size_t)MTOK * 5 * 2);
  __bf16* wq    = (__bf16*)alloc((size_t)NLAYER * 128 * 384 * 2);  // [l][384][128]
  __bf16* wo    = (__bf16*)alloc((size_t)NLAYER * 128 * 128 * 2);  // [l][128][128]
  __bf16* w1    = (__bf16*)alloc((size_t)NLAYER * 128 * 512 * 2);  // [l][512][128]
  __bf16* w2    = (__bf16*)alloc((size_t)NLAYER * 512 * 128 * 2);  // [l][128][512]
  __bf16* wE2   = (__bf16*)alloc(64 * 128 * 2);                    // [128][64]
  __bf16* wE3   = (__bf16*)alloc(128 * 128 * 2);                   // [128][128]
  __bf16* wR1   = (__bf16*)alloc(128 * 128 * 2);                   // [128][128]
  __bf16* wR2   = (__bf16*)alloc(128 * 64 * 2);                    // [64][128]
  __bf16* wR3   = (__bf16*)alloc(64 * 5 * 2);                      // [5][64]
  __bf16* wC    = (__bf16*)alloc((size_t)1500 * 2500 * 2);         // [2500][1500]
  __bf16* posk  = (__bf16*)alloc((size_t)NLAYER * 4096 * 2);
  __bf16* posq  = (__bf16*)alloc((size_t)NLAYER * 4096 * 2);

  auto convT = [&](const float* s, __bf16* d, int K, int N, int L) {
    long n = (long)L * K * N;
    convT_kernel<<<(unsigned)((n + 255) / 256), 256, 0, stream>>>(s, d, K, N, L);
  };
  auto gemm = [&](const __bf16* A, int lda, const __bf16* Wt, int ldw,
                  const float* bias, const float* resid, int ldr,
                  const float* lns, const float* lnb,
                  float* oF, int ldoF, __bf16* oB, int ldoB,
                  int M, int N, int K, int flags) {
    dim3 g((M + 127) / 128, (N + 127) / 128);
    gemm_bf16_kernel<<<g, 256, 0, stream>>>(A, lda, Wt, ldw, bias, resid, ldr,
                                            lns, lnb, oF, ldoF, oB, ldoB,
                                            M, N, K, flags);
  };

  convT(qkv_W,      wq,  128, 384, NLAYER);
  convT(attn_out_W, wo,  128, 128, NLAYER);
  convT(ffn_W1,     w1,  128, 512, NLAYER);
  convT(ffn_W2,     w2,  512, 128, NLAYER);
  convT(emb_W2,     wE2, 64,  128, 1);
  convT(emb_W3,     wE3, 128, 128, 1);
  convT(rec_W1,     wR1, 128, 128, 1);
  convT(rec_W2,     wR2, 128, 64,  1);
  convT(rec_W3,     wR3, 64,  5,   1);
  convT(conn_W,     wC,  1500, 2500, 1);
  posqk_kernel<<<(NLAYER * 2 * 16 * 128 + 255) / 256, 256, 0, stream>>>(
      rel_emb, pos_k_W, pos_q_W, pos_q_b, posk, posq);

  embed_first_kernel<<<(MTOK * 64 + 255) / 256, 256, 0, stream>>>(
      inputs, emb_W1, emb_b1, e1);
  gemm(e1, 64,  wE2, 64,  emb_b2, nullptr, 0, nullptr, nullptr,
       nullptr, 0, e2, 128, MTOK, 128, 64, 1 | 2);
  gemm(e2, 128, wE3, 128, emb_b3, nullptr, 0, nullptr, nullptr,
       e3, 128, nullptr, 0, MTOK, 128, 128, 1 | 2);
  embed_ln_kernel<<<(MTOK + 255) / 256, 256, 0, stream>>>(
      e3, inputs, tok_emb, emb_ln_s, emb_ln_b, mask, x_f32, x_bf);

  for (int l = 0; l < NLAYER; ++l) {
    gemm(x_bf, 128, wq + (size_t)l * 128 * 384, 128, nullptr, nullptr, 0,
         nullptr, nullptr, qkv, 384, nullptr, 0, MTOK, 384, 128, 0);
    attn_kernel<<<dim3(19, BATCH * NHEAD), 32, 0, stream>>>(
        qkv, q_bias + l * 128, v_bias + l * 128,
        posk + (size_t)l * 4096, posq + (size_t)l * 4096, mask, ctx);
    gemm(ctx, 128, wo + (size_t)l * 128 * 128, 128, attn_out_b + l * 128,
         x_f32, 128, ln1_s + l * 128, ln1_b + l * 128,
         x_f32, 128, x_bf, 128, MTOK, 128, 128, 1 | 4);
    gemm(x_bf, 128, w1 + (size_t)l * 128 * 512, 128, ffn_b1 + l * 512,
         nullptr, 0, nullptr, nullptr,
         nullptr, 0, hmid, 512, MTOK, 512, 128, 1 | 2);
    gemm(hmid, 512, w2 + (size_t)l * 512 * 128, 512, ffn_b2 + l * 128,
         x_f32, 128, ln2_s + l * 128, ln2_b + l * 128,
         x_f32, 128, x_bf, 128, MTOK, 128, 512, 1 | 4);
  }

  gemm(x_bf, 128, wR1, 128, rec_b1, nullptr, 0, nullptr, nullptr,
       nullptr, 0, r1, 128, MTOK, 128, 128, 1 | 2);
  gemm(r1, 128, wR2, 128, rec_b2, nullptr, 0, nullptr, nullptr,
       nullptr, 0, r2, 64, MTOK, 64, 128, 1 | 2);
  gemm(r2, 64, wR3, 64, rec_b3, nullptr, 0, nullptr, nullptr,
       nullptr, 0, r3, 5, MTOK, 5, 64, 1 | 2);
  // r3 is [B*S,5] contiguous == [64,1500]
  gemm(r3, 1500, wC, 1500, conn_b, nullptr, 0, nullptr, nullptr,
       out, 2500, nullptr, 0, BATCH, 2500, 1500, 1);
}
